// GNNClassifier_30107720745623
// MI455X (gfx1250) — matrix operand order repaired and checked by reading, not compiled, not run on hardware
//
#include <hip/hip_runtime.h>
#include <stdint.h>

// ---------------------------------------------------------------------------
// GNN classifier for MI455X (gfx1250): bf16 WMMA GEMMs (pre-swizzled weights)
// + L2-resident edge scatter.
// ---------------------------------------------------------------------------

typedef __attribute__((ext_vector_type(16))) __bf16          v16bf;
typedef __attribute__((ext_vector_type(16))) unsigned short  v16u16;
typedef __attribute__((ext_vector_type(8)))  float           v8f;
typedef __attribute__((ext_vector_type(8)))  unsigned short  ushort8;

static constexpr int   kIn = 128, kH1 = 256, kH2 = 128, kH3 = 256, kOut = 4;
static constexpr int   kLayers = 4;
static constexpr float kKeep = 0.9f;

__device__ __forceinline__ unsigned short f32_to_bf16_bits(float f) {
  unsigned int u = __builtin_bit_cast(unsigned int, f);
  u += 0x7FFFu + ((u >> 16) & 1u);          // round-to-nearest-even
  return (unsigned short)(u >> 16);
}

__global__ void k_f32_to_bf16(const float* __restrict__ in,
                              unsigned short* __restrict__ out, int n) {
  int i = blockIdx.x * blockDim.x + threadIdx.x;
  if (i < n) out[i] = f32_to_bf16_bits(in[i]);
}

__global__ void k_zero_f32(float* __restrict__ p, int n) {
  int i = blockIdx.x * blockDim.x + threadIdx.x;
  if (i < n) p[i] = 0.0f;
}

__global__ void k_degree(const int* __restrict__ dst, float* __restrict__ deg, int nE) {
  int e = blockIdx.x * blockDim.x + threadIdx.x;
  if (e < nE) atomicAdd(&deg[dst[e]], 1.0f);
}

__global__ void k_dinv(float* __restrict__ deg, int n) {
  int i = blockIdx.x * blockDim.x + threadIdx.x;
  if (i < n) deg[i] = rsqrtf(deg[i] + 1.0f);   // +1 self-loop => deg > 0 always
}

// ---------------------------------------------------------------------------
// Pack weight W (f32, [K,N] row-major) into the WMMA bf16 B-operand layout,
// zero-padded to ceil(N/16)*16 columns:
//   Bsw[((tc*K32 + blk)*32 + lane)*16 + e]
//     col = tc*16 + (lane&15),  k = blk*32 + 16*(e>>3) + 8*(lane>>4) + (e&7)
// Each GEMM lane then reads its whole B fragment as one contiguous 32B load.
// ---------------------------------------------------------------------------
__global__ void k_pack_B(const float* __restrict__ W, unsigned short* __restrict__ Bsw,
                         int K, int N, int total) {
  int i = blockIdx.x * blockDim.x + threadIdx.x;
  if (i >= total) return;
  int e    = i & 15;
  int lane = (i >> 4) & 31;
  int rest = i >> 9;
  int K32  = K >> 5;
  int blk  = rest % K32;
  int tc   = rest / K32;
  int col  = (tc << 4) + (lane & 15);
  int k    = (blk << 5) + ((e >> 3) << 4) + ((lane >> 4) << 3) + (e & 7);
  Bsw[i] = (col < N) ? f32_to_bf16_bits(W[(size_t)k * N + col]) : (unsigned short)0;
}

// ---------------------------------------------------------------------------
// GEMM: C = act(A[M,K](bf16) @ B[K,N] + bias), one 16x16 tile per wave,
// K-loop step 32 via v_wmma_f32_16x16x32_bf16.  B comes pre-swizzled (k_pack_B).
// A layout per ISA: lane = two contiguous 8-half K-chunks -> two 16B loads.
// act: 0 = none, 1 = relu, 2 = sigmoid.  M must be a multiple of 16.
// ---------------------------------------------------------------------------
__global__ void __launch_bounds__(256)
k_gemm_bf16(const unsigned short* __restrict__ A,
            const unsigned short* __restrict__ Bsw,
            const float* __restrict__ bias,
            int M, int K, int N, int act,
            unsigned short* __restrict__ out_bf,
            float* __restrict__ out_f32) {
  int lane  = threadIdx.x & 31;
  int wid   = blockIdx.x * (blockDim.x >> 5) + (threadIdx.x >> 5);
  int tn    = (N + 15) >> 4;
  int tiles = (M >> 4) * tn;
  if (wid >= tiles) return;                 // wave-uniform: EXEC stays all-ones
  int tm = wid / tn;
  int tc = wid - tm * tn;
  int row0 = tm << 4;
  int half = lane >> 4, l15 = lane & 15;
  int K32  = K >> 5;

  const unsigned short* Arow  = A + (size_t)(row0 + l15) * K + 8 * half;
  const unsigned short* Blane = Bsw + ((size_t)tc * K32 * 32 + lane) * 16;

  v8f acc = {};
  for (int blk = 0; blk < K32; ++blk) {
    int k0 = blk << 5;
    ushort8 alo = *(const ushort8*)(Arow + k0);        // K = k0 + 8*half + [0..7]
    ushort8 ahi = *(const ushort8*)(Arow + k0 + 16);   // K = k0 + 16 + 8*half + [0..7]
    v16u16  au  = __builtin_shufflevector(alo, ahi,
                    0, 1, 2, 3, 4, 5, 6, 7, 8, 9, 10, 11, 12, 13, 14, 15);
    v16u16  bu  = *(const v16u16*)(Blane + (size_t)blk * 512);  // contiguous 32B
    v16bf a = __builtin_bit_cast(v16bf, au);
    v16bf b = __builtin_bit_cast(v16bf, bu);
    acc = __builtin_amdgcn_wmma_f32_16x16x32_bf16(
        /*neg_a=*/false, a, /*neg_b=*/false, b,
        /*c_mod=*/(short)0, acc, /*reuse_a=*/false, /*reuse_b=*/false);
  }
  int col = (tc << 4) + l15;
  if (col < N) {
    float bv = bias ? bias[col] : 0.0f;
#pragma unroll
    for (int j = 0; j < 8; ++j) {
      int   row = row0 + j + 8 * half;                 // D: row = j + 8*half
      float v   = acc[j] + bv;
      if (act == 1)      v = fmaxf(v, 0.0f);
      else if (act == 2) v = 1.0f / (1.0f + __expf(-v));
      size_t idx = (size_t)row * N + col;
      if (out_bf)  out_bf[idx]  = f32_to_bf16_bits(v);
      if (out_f32) out_f32[idx] = v;
    }
  }
}

// agg[i,f] = m[i,f] * dinv[i]^2  (self-loop contribution, also initializes agg)
__global__ void k_self_init(const float* __restrict__ m, const float* __restrict__ dinv,
                            float* __restrict__ agg, int n) {
  int t = blockIdx.x * blockDim.x + threadIdx.x;
  if (t < n) {
    float di = dinv[t >> 7];
    agg[t] = m[t] * di * di;
  }
}

// agg[dst] += m[src] * dinv[src]*dinv[dst]; 32 lanes x float4 per edge.
// agg (51 MB) is L2-resident on MI455X (192 MB L2) -> atomics resolve in L2.
__global__ void k_scatter(const int* __restrict__ src, const int* __restrict__ dst,
                          const float* __restrict__ dinv, const float* __restrict__ m,
                          float* __restrict__ agg, int nE) {
  long long t = (long long)blockIdx.x * blockDim.x + threadIdx.x;
  int e = (int)(t >> 5);
  if (e >= nE) return;
  int f = ((int)t & 31) << 2;
  int s = src[e], d = dst[e];
  float w = dinv[s] * dinv[d];
  const float4 mv = *(const float4*)(m + (size_t)s * 128 + f);
  float* ap = agg + (size_t)d * 128 + f;
  atomicAdd(ap + 0, mv.x * w);
  atomicAdd(ap + 1, mv.y * w);
  atomicAdd(ap + 2, mv.z * w);
  atomicAdd(ap + 3, mv.w * w);
}

// h = bf16(dropout(relu(agg + bias)))  — deterministic hash-based dropout.
__global__ void k_post_agg(const float* __restrict__ agg, const float* __restrict__ bias,
                           unsigned short* __restrict__ h_bf, int n, unsigned int seed) {
  int t = blockIdx.x * blockDim.x + threadIdx.x;
  if (t >= n) return;
  float v = agg[t] + bias[t & 127];
  v = fmaxf(v, 0.0f);
  unsigned int x = (unsigned int)t * 0x9E3779B9u ^ seed;
  x ^= x >> 16; x *= 0x7feb352du; x ^= x >> 15; x *= 0x846ca68bu; x ^= x >> 16;
  bool keep = (x >> 8) < (unsigned int)(kKeep * 16777216.0f);
  v = keep ? v * (1.0f / kKeep) : 0.0f;
  h_bf[t] = f32_to_bf16_bits(v);
}

// ---------------------------------------------------------------------------
extern "C" void kernel_launch(void* const* d_in, const int* in_sizes, int n_in,
                              void* d_out, int out_size, void* d_ws, size_t ws_size,
                              hipStream_t stream) {
  (void)n_in; (void)out_size; (void)ws_size;
  const float* x  = (const float*)d_in[0];
  const int*   ei = (const int*)d_in[1];     // [2,E]: row0 = src, row1 = dst
  // d_in[2] = edge_attr (unused, matches reference)
  const float* W1 = (const float*)d_in[3];
  const float* b1 = (const float*)d_in[4];
  const float* W2 = (const float*)d_in[5];
  const float* b2 = (const float*)d_in[6];
  const float* gW = (const float*)d_in[7];   // [4,128,128]
  const float* gb = (const float*)d_in[8];   // [4,128]
  const float* W3 = (const float*)d_in[9];
  const float* b3 = (const float*)d_in[10];
  const float* W4 = (const float*)d_in[11];
  const float* b4 = (const float*)d_in[12];
  float* out = (float*)d_out;

  const int N = in_sizes[0] / kIn;   // 100000 (multiple of 16)
  const int E = in_sizes[1] / 2;     // 1600000
  const int* src = ei;
  const int* dst = ei + E;

  // Packed-B sizes: ceil(N/16)*16 * K bf16 elements.
  auto packedElems = [](int K, int Nc) {
    return (size_t)((Nc + 15) / 16) * (K >> 5) * 512;
  };

  // Workspace carve-out (256B aligned).
  char* ws = (char*)d_ws;
  size_t off = 0;
  auto alloc = [&](size_t bytes) -> void* {
    void* p = ws + off;
    off = (off + bytes + 255) & ~(size_t)255;
    return p;
  };
  float*          dinv = (float*)alloc((size_t)N * 4);
  unsigned short* B0   = (unsigned short*)alloc((size_t)N * kH2 * 2); // x_bf16 / h_bf16 [N,128]
  unsigned short* B1   = (unsigned short*)alloc((size_t)N * kH1 * 2); // [N,256] bf16
  float*          Mf   = (float*)alloc((size_t)N * kH2 * 4);          // m = h@W  (f32)
  float*          Ag   = (float*)alloc((size_t)N * kH2 * 4);          // agg      (f32)
  unsigned short* W1s  = (unsigned short*)alloc(packedElems(kIn, kH1) * 2);
  unsigned short* W2s  = (unsigned short*)alloc(packedElems(kH1, kH2) * 2);
  unsigned short* Wgs  = (unsigned short*)alloc((size_t)kLayers * packedElems(kH2, kH2) * 2);
  unsigned short* W3s  = (unsigned short*)alloc(packedElems(kH2, kH3) * 2);
  unsigned short* W4s  = (unsigned short*)alloc(packedElems(kH3, kOut) * 2);

  auto cdiv = [](long long a, long long b) { return (int)((a + b - 1) / b); };

  auto packB = [&](const float* W, unsigned short* Bsw, int K, int Nc) {
    int total = (int)packedElems(K, Nc);
    k_pack_B<<<cdiv(total, 256), 256, 0, stream>>>(W, Bsw, K, Nc, total);
  };

  // Pre-swizzle all weights into WMMA B-operand layout (tiny, deterministic).
  packB(W1, W1s, kIn, kH1);
  packB(W2, W2s, kH1, kH2);
  for (int l = 0; l < kLayers; ++l)
    packB(gW + (size_t)l * kH2 * kH2, Wgs + (size_t)l * packedElems(kH2, kH2), kH2, kH2);
  packB(W3, W3s, kH2, kH3);
  packB(W4, W4s, kH3, kOut);

  // x -> bf16
  k_f32_to_bf16<<<cdiv((long long)N * kIn, 256), 256, 0, stream>>>(x, B0, N * kIn);
  // GCN degree normalization: dinv = rsqrt(in-degree + 1)
  k_zero_f32<<<cdiv(N, 256), 256, 0, stream>>>(dinv, N);
  k_degree<<<cdiv(E, 256), 256, 0, stream>>>(dst, dinv, E);
  k_dinv<<<cdiv(N, 256), 256, 0, stream>>>(dinv, N);

  auto gemm = [&](const unsigned short* A, const unsigned short* Bsw, const float* bias,
                  int K, int Nc, int act, unsigned short* obf, float* of32) {
    int tn = (Nc + 15) / 16;
    long long tiles = (long long)(N / 16) * tn;   // one wave per 16x16 tile
    k_gemm_bf16<<<cdiv(tiles, 8), 256, 0, stream>>>(A, Bsw, bias, N, K, Nc, act, obf, of32);
  };

  // Encoder MLP
  gemm(B0, W1s, b1, kIn, kH1, /*relu*/1, B1, nullptr);   // [N,256]
  gemm(B1, W2s, b2, kH1, kH2, /*relu*/1, B0, nullptr);   // [N,128]

  // GCN layers
  for (int l = 0; l < kLayers; ++l) {
    gemm(B0, Wgs + (size_t)l * packedElems(kH2, kH2), nullptr, kH2, kH2, /*none*/0,
         nullptr, Mf);
    k_self_init<<<cdiv((long long)N * kH2, 256), 256, 0, stream>>>(Mf, dinv, Ag, N * kH2);
    k_scatter<<<cdiv((long long)E * 32, 256), 256, 0, stream>>>(src, dst, dinv, Mf, Ag, E);
    k_post_agg<<<cdiv((long long)N * kH2, 256), 256, 0, stream>>>(
        Ag, gb + (size_t)l * kH2, B0, N * kH2, 0x9E3779B9u * (unsigned)(l + 1));
  }

  // Decoder MLP
  gemm(B0, W3s, b3, kH2, kH3, /*relu*/1, B1, nullptr);       // [N,256]
  gemm(B1, W4s, b4, kH3, kOut, /*sigmoid*/2, nullptr, out);  // [N,4] f32
}
